// GNNModel_71279277244837
// MI455X (gfx1250) — matrix operand order repaired and checked by reading, not compiled, not run on hardware
//
#include <hip/hip_runtime.h>
#include <hip/hip_bf16.h>

// CDNA5 / gfx1250. wave32. H (hidden) = 32 fixed by the reference model.
#define HID 32

typedef __attribute__((ext_vector_type(16))) __bf16 v16bf;
typedef __attribute__((ext_vector_type(8)))  float  v8f;

static __device__ __forceinline__ __bf16 f2bf(float f) {
    // round-to-nearest-even float32 -> bf16
    unsigned u = __float_as_uint(f);
    unsigned r = (u + 0x7FFFu + ((u >> 16) & 1u)) >> 16;
    unsigned short s = (unsigned short)r;
    __bf16 o;
    __builtin_memcpy(&o, &s, sizeof(o));
    return o;
}

// ---------------------------------------------------------------------------
// 0) zero all accumulators used this call (deterministic; ws is poisoned once)
// ---------------------------------------------------------------------------
__global__ void k_init(float* __restrict__ agg, int* __restrict__ deg,
                       float* __restrict__ a1, float* __restrict__ pool,
                       int* __restrict__ cnt, int n, int b) {
    long long i = (long long)blockIdx.x * blockDim.x + threadIdx.x;
    if (i < (long long)n * HID) agg[i] = 0.0f;
    if (i < n) { deg[i] = 0; a1[i] = 0.0f; }
    if (i < b) { pool[i] = 0.0f; cnt[i] = 0; }
}

// ---------------------------------------------------------------------------
// 1) in-degree over dst (self loop added analytically later)
// ---------------------------------------------------------------------------
__global__ void k_deg(const int* __restrict__ dst, int* __restrict__ deg, int e) {
    int i = blockIdx.x * blockDim.x + threadIdx.x;
    if (i < e) atomicAdd(&deg[dst[i]], 1);
}

// 2) dinv = (deg + 1)^-1/2   (deg+1 >= 1, so the where() in the ref is moot)
__global__ void k_dinv(const int* __restrict__ deg, float* __restrict__ dinv, int n) {
    int i = blockIdx.x * blockDim.x + threadIdx.x;
    if (i < n) dinv[i] = rsqrtf((float)(deg[i] + 1));
}

// ---------------------------------------------------------------------------
// 3) layer-1 scatter: x is [N,1] so the message is a SCALAR per edge.
//    a1[d] += x[s] * dinv[s] * dinv[d]
// ---------------------------------------------------------------------------
__global__ void k_edge1(const int* __restrict__ src, const int* __restrict__ dst,
                        const float* __restrict__ x, const float* __restrict__ dinv,
                        float* __restrict__ a1, int e) {
    int i = blockIdx.x * blockDim.x + threadIdx.x;
    if (i < e) {
        int s = src[i], d = dst[i];
        atomicAdd(&a1[d], x[s] * dinv[s] * dinv[d]);
    }
}

// ---------------------------------------------------------------------------
// 4) h1[n][c] = relu((a1[n] + x[n]*dinv[n]^2) * W1[c] + b1[c])
//    also counts nodes per graph for the mean pool
// ---------------------------------------------------------------------------
__global__ void k_h1(const float* __restrict__ x, const float* __restrict__ a1,
                     const float* __restrict__ dinv, const float* __restrict__ W1,
                     const float* __restrict__ b1, const int* __restrict__ batch,
                     float* __restrict__ h1, int* __restrict__ cnt, int n) {
    long long i = (long long)blockIdx.x * blockDim.x + threadIdx.x;
    if (i < (long long)n * HID) {
        int node = (int)(i >> 5);
        int c    = (int)(i & 31);
        float di = dinv[node];
        float v  = (a1[node] + x[node] * di * di) * W1[c] + b1[c];
        h1[i] = v > 0.0f ? v : 0.0f;
        if (c == 0) atomicAdd(&cnt[batch[node]], 1);
    }
}

// ---------------------------------------------------------------------------
// 5) layer-2 scatter (matmul pulled AFTER aggregation; scatter-add is linear):
//    agg[d][c] += h1[s][c] * dinv[s]*dinv[d]
//    one wave32 per edge: lane = channel -> fully coalesced gather + atomic
// ---------------------------------------------------------------------------
__global__ void k_edge2(const int* __restrict__ src, const int* __restrict__ dst,
                        const float* __restrict__ h1, const float* __restrict__ dinv,
                        float* __restrict__ agg, int e) {
    long long i = (long long)blockIdx.x * blockDim.x + threadIdx.x;
    if (i < (long long)e * HID) {
        int eidx = (int)(i >> 5);
        int c    = (int)(i & 31);
        int s = src[eidx], d = dst[eidx];
        float w = dinv[s] * dinv[d];
        atomicAdd(&agg[(long long)d * HID + c], h1[(long long)s * HID + c] * w);
    }
}

// ---------------------------------------------------------------------------
// 6) WMMA tile kernel: per 16-node tile,
//      agg2 = agg + dinv^2 * h1                    (self loop)
//      h2   = relu(agg2 @ W2 + b2)                 (2x v_wmma_f32_16x16x32_bf16)
//      pool[batch[n]] += h2[n] . Wl                (fused final linear)
//    One wave per block, exact grid => EXEC all-1s at every WMMA.
// ---------------------------------------------------------------------------
__global__ void __launch_bounds__(32)
k_gemm2(const float* __restrict__ agg, const float* __restrict__ h1,
        const float* __restrict__ dinv, const float* __restrict__ W2,
        const float* __restrict__ b2, const float* __restrict__ Wl,
        const int* __restrict__ batch, float* __restrict__ pool, int n) {
    const int lane = threadIdx.x;       // 0..31
    const int row  = lane & 15;         // M (A-row / C-column index)
    const int half = lane >> 4;         // lane half selects K sub-block
    const int base = blockIdx.x * 16;

    int node = base + row;
    if (node >= n) node = n - 1;        // clamp loads; writes guarded below
    const float d2 = dinv[node] * dinv[node];
    const float* __restrict__ aRow = agg + (long long)node * HID;
    const float* __restrict__ hRow = h1  + (long long)node * HID;

    // A (16x32 bf16), documented layout: lane<16 holds K {0..7,16..23},
    // lane>=16 holds K {8..15,24..31}:  k(i) = i + (i&8) + 8*half
    v16bf a;
#pragma unroll
    for (int i = 0; i < 16; ++i) {
        int k = i + (i & 8) + 8 * half;
        a[i] = f2bf(aRow[k] + d2 * hRow[k]);
    }

    // B (32x16 bf16): lane = output column, same K order as A: k(i) = i + (i&8) + 8*half
    v16bf bA, bB;
#pragma unroll
    for (int i = 0; i < 16; ++i) {
        int k = i + (i & 8) + 8 * half;
        bA[i] = f2bf(W2[k * HID + row]);        // columns  0..15
        bB[i] = f2bf(W2[k * HID + 16 + row]);   // columns 16..31
    }

    v8f c0 = {};
    v8f c1 = {};
    c0 = __builtin_amdgcn_wmma_f32_16x16x32_bf16(false, a, false, bA,
                                                 (short)0, c0, false, false);
    c1 = __builtin_amdgcn_wmma_f32_16x16x32_bf16(false, a, false, bB,
                                                 (short)0, c1, false, false);

    // C/D layout: VGPR j, lanes 0-15 -> row j, lanes 16-31 -> row j+8; col = row.
    const int   n0 = row, n1 = 16 + row;
    const float w0 = Wl[n0], w1 = Wl[n1];
    const float q0 = b2[n0], q1 = b2[n1];

#pragma unroll
    for (int j = 0; j < 8; ++j) {
        float p0 = c0[j] + q0; p0 = p0 > 0.0f ? p0 : 0.0f;
        float p1 = c1[j] + q1; p1 = p1 > 0.0f ? p1 : 0.0f;
        float part = p0 * w0 + p1 * w1;           // partial h2 . Wl
        // reduce the 16 columns held by this lane-half (xor of bits 0..3
        // stays within the half)
        part += __shfl_xor(part, 1, 32);
        part += __shfl_xor(part, 2, 32);
        part += __shfl_xor(part, 4, 32);
        part += __shfl_xor(part, 8, 32);
        int outNode = base + j + 8 * half;
        if (row == 0 && outNode < n)
            atomicAdd(&pool[batch[outNode]], part);
    }
}

// 7) out[b] = pool[b] / max(cnt[b], 1) + bl
__global__ void k_final(const float* __restrict__ pool, const int* __restrict__ cnt,
                        const float* __restrict__ bl, float* __restrict__ out, int b) {
    int i = blockIdx.x * blockDim.x + threadIdx.x;
    if (i < b) {
        float c = (float)cnt[i];
        c = c > 1.0f ? c : 1.0f;
        out[i] = pool[i] / c + bl[0];
    }
}

// ---------------------------------------------------------------------------
extern "C" void kernel_launch(void* const* d_in, const int* in_sizes, int n_in,
                              void* d_out, int out_size, void* d_ws, size_t ws_size,
                              hipStream_t stream) {
    const float* x     = (const float*)d_in[0];   // [N,1]
    const int*   ei    = (const int*)  d_in[1];   // [2,E]
    const int*   batch = (const int*)  d_in[2];   // [N]
    const float* W1    = (const float*)d_in[3];   // [1,32]
    const float* b1    = (const float*)d_in[4];   // [32]
    const float* W2    = (const float*)d_in[5];   // [32,32]
    const float* b2    = (const float*)d_in[6];   // [32]
    const float* Wl    = (const float*)d_in[7];   // [32,1]
    const float* bl    = (const float*)d_in[8];   // [1]
    float* out = (float*)d_out;

    const int n = in_sizes[0];          // N
    const int e = in_sizes[1] / 2;      // E
    const int b = out_size;             // B

    const int* src = ei;
    const int* dst = ei + e;

    // carve workspace
    char* ws = (char*)d_ws;
    size_t o = 0;
    auto al = [](size_t v) { return (v + 255) & ~(size_t)255; };
    int*   deg  = (int*)  (ws + o); o = al(o + (size_t)n * 4);
    float* dinv = (float*)(ws + o); o = al(o + (size_t)n * 4);
    float* a1   = (float*)(ws + o); o = al(o + (size_t)n * 4);
    float* h1   = (float*)(ws + o); o = al(o + (size_t)n * HID * 4);
    float* agg  = (float*)(ws + o); o = al(o + (size_t)n * HID * 4);
    float* pool = (float*)(ws + o); o = al(o + (size_t)b * 4);
    int*   cnt  = (int*)  (ws + o); o = al(o + (size_t)b * 4);
    (void)ws_size; (void)n_in;

    const int T = 256;
    long long nh = (long long)n * HID;
    long long eh = (long long)e * HID;

    k_init <<<(unsigned)((nh + T - 1) / T), T, 0, stream>>>(agg, deg, a1, pool, cnt, n, b);
    k_deg  <<<(e + T - 1) / T, T, 0, stream>>>(dst, deg, e);
    k_dinv <<<(n + T - 1) / T, T, 0, stream>>>(deg, dinv, n);
    k_edge1<<<(e + T - 1) / T, T, 0, stream>>>(src, dst, x, dinv, a1, e);
    k_h1   <<<(unsigned)((nh + T - 1) / T), T, 0, stream>>>(x, a1, dinv, W1, b1, batch, h1, cnt, n);
    k_edge2<<<(unsigned)((eh + T - 1) / T), T, 0, stream>>>(src, dst, h1, dinv, agg, e);
    k_gemm2<<<(n + 15) / 16, 32, 0, stream>>>(agg, h1, dinv, W2, b2, Wl, batch, pool, n);
    k_final<<<(b + T - 1) / T, T, 0, stream>>>(pool, cnt, bl, out, b);
}
